// LSTMLayer_84670985273417
// MI455X (gfx1250) — compile-verified, matrix operand order checked by reading, not run on hardware
//
#include <hip/hip_runtime.h>

// ---------------------------------------------------------------------------
// LSTM layer for MI455X (gfx1250, wave32, WMMA).
//   B=64, S=2048, E=H=256.
// Phase 1: pre[t] = x[:,t,:] @ Wx(4 gates) + bias  -> bf16, WMMA fragment layout
// Phase 2: persistent 32-WG scan; Wh slice + h in LDS, c in VGPRs,
//          one global atomic barrier per timestep, WMMA bf16 step-GEMM.
// Workspace needed: ~258 MiB.
// ---------------------------------------------------------------------------

typedef __attribute__((ext_vector_type(16))) __bf16          v16bf;
typedef __attribute__((ext_vector_type(8)))  __bf16          v8bf;
typedef __attribute__((ext_vector_type(8)))  float           v8f;
typedef __attribute__((ext_vector_type(8)))  unsigned short  us8;

#define CAT16(lo, hi) __builtin_shufflevector((lo), (hi), 0,1,2,3,4,5,6,7,8,9,10,11,12,13,14,15)

__device__ __forceinline__ unsigned short f2bf(float f) {
  unsigned u = __float_as_uint(f);
  u += 0x7FFFu + ((u >> 16) & 1u);               // round-to-nearest-even
  return (unsigned short)(u >> 16);
}
__device__ __forceinline__ float bf2f(unsigned short h) {
  return __uint_as_float(((unsigned)h) << 16);
}
__device__ __forceinline__ float sigf(float x)  { return 1.0f / (1.0f + __expf(-x)); }
__device__ __forceinline__ float tanhf_(float x){ return 1.0f - 2.0f / (__expf(2.0f * x) + 1.0f); }

// ---------------------------------------------------------------------------
// Kernel 0: convert W (fp32 [512,256] per gate) -> bf16 Wx[4][256][256],
//           Wh[4][256][256]; also resets the scan barrier counter.
// ---------------------------------------------------------------------------
__global__ __launch_bounds__(256) void lstm_convw(
    const float* __restrict__ Wf, const float* __restrict__ Wi,
    const float* __restrict__ Wc, const float* __restrict__ Wo,
    unsigned short* __restrict__ wx, unsigned short* __restrict__ wh,
    unsigned int* __restrict__ counter)
{
  const size_t gid = (size_t)blockIdx.x * 256 + threadIdx.x;   // 524288 total
  if (gid == 0) *counter = 0u;                                  // per-launch reset
  const int g = (int)(gid >> 17);
  const int rem = (int)(gid & 131071);
  const int k = rem >> 8, n = rem & 255;
  const float* W = (g == 0) ? Wf : (g == 1) ? Wi : (g == 2) ? Wc : Wo;
  const unsigned short v = f2bf(W[(size_t)k * 256 + n]);
  if (k < 256) wx[((size_t)g * 256 + k) * 256 + n] = v;
  else         wh[((size_t)g * 256 + (k - 256)) * 256 + n] = v;
}

// ---------------------------------------------------------------------------
// Kernel 1: pre-GEMM.  grid = (16 col-chunks of 64, 2 batch-halves of 32, 2048 t)
// 128 threads = 4 waves; each wave owns 1 M-tile x 2 N-tiles, K=256 (8 WMMA k-steps).
// pre stored bf16 in C-fragment layout:
//   idx = (((t*4 + mtile)*64 + ctile)*32 + lane)*8 + r
// ---------------------------------------------------------------------------
__global__ __launch_bounds__(128) void lstm_pregemm(
    const float* __restrict__ x, const unsigned short* __restrict__ wx,
    const float* __restrict__ b_f, const float* __restrict__ b_i,
    const float* __restrict__ b_c, const float* __restrict__ b_o,
    unsigned short* __restrict__ pre)
{
  __shared__ alignas(16) unsigned short Alds[32][264];   // x tile, bf16, padded
  __shared__ alignas(16) unsigned short Blds[64][264];   // Wx^T slice (col-major)
  const int c64 = blockIdx.x, mhalf = blockIdx.y, t = blockIdx.z;
  const int tid = threadIdx.x;

  // Stage A: x[b, t, 0:256] for b in [mhalf*32, +32), fp32 -> bf16
  for (int u = tid; u < 1024; u += 128) {                // 1024 units of 8 elems
    const int row = u >> 5, off = (u & 31) * 8;
    const float* s = x + ((size_t)(mhalf * 32 + row) * 2048 + t) * 256 + off;
    us8 v;
#pragma unroll
    for (int r = 0; r < 8; ++r) v[r] = f2bf(s[r]);
    *(us8*)&Alds[row][off] = v;
  }
  // Stage B transposed: Wx[g][k][n] -> Blds[c][k], cols c64*64..+63
  {
    const unsigned short* s = wx + ((size_t)(c64 >> 2)) * 65536 + (c64 & 3) * 64;
    for (int i = 0; i < 16; ++i) {
      const int unit = tid + 128 * i;                    // 2048 units of 8
      const int cb = (unit & 7) * 8, k = unit >> 3;
      us8 v = *(const us8*)&s[(size_t)k * 256 + cb];
#pragma unroll
      for (int r = 0; r < 8; ++r) Blds[cb + r][k] = v[r];
    }
  }
  __syncthreads();

  const int w = tid >> 5, lane = tid & 31;
  const int mt = w & 1, nt0 = (w >> 1) * 2;
  const int l15 = lane & 15, k8 = ((lane >> 4) & 1) * 8;
  v8f acc[2] = {{}, {}};
  const unsigned short* ap  = &Alds[mt * 16 + l15][0];
  const unsigned short* b0p = &Blds[nt0 * 16 + l15][0];
  const unsigned short* b1p = &Blds[nt0 * 16 + 16 + l15][0];
#pragma unroll
  for (int kt = 0; kt < 8; ++kt) {
    const int ka = kt * 32 + k8;
    v16bf a  = CAT16(*(const v8bf*)&ap[ka],  *(const v8bf*)&ap[ka + 16]);
    v16bf b0 = CAT16(*(const v8bf*)&b0p[ka], *(const v8bf*)&b0p[ka + 16]);
    v16bf b1 = CAT16(*(const v8bf*)&b1p[ka], *(const v8bf*)&b1p[ka + 16]);
    acc[0] = __builtin_amdgcn_wmma_f32_16x16x32_bf16(false, a, false, b0, (short)0, acc[0], false, false);
    acc[1] = __builtin_amdgcn_wmma_f32_16x16x32_bf16(false, a, false, b1, (short)0, acc[1], false, false);
  }
  const int mtg = mhalf * 2 + mt;
#pragma unroll
  for (int j = 0; j < 2; ++j) {
    const int ctile = c64 * 4 + nt0 + j;
    const int colg  = ctile * 16 + l15;
    const int g = colg >> 8, n = colg & 255;
    const float* bp = (g == 0) ? b_f : (g == 1) ? b_i : (g == 2) ? b_c : b_o;
    const float bias = bp[n];
    us8 o;
#pragma unroll
    for (int r = 0; r < 8; ++r) o[r] = f2bf(acc[j][r] + bias);
    *(us8*)&pre[((((size_t)t * 4 + mtg) * 64 + ctile) * 32 + lane) * 8] = o;
  }
}

// ---------------------------------------------------------------------------
// Kernel 2: persistent scan. 32 WGs x 256 threads (8 waves).
// WG w: gate g=w/8, column slice oct=w%8 (32 cols). Per step:
//   G[g][:, slice] = h @ Wh_slice + pre  (WMMA, A/B from LDS)
//   -> global barrier (atomic counter) -> redundant elementwise update,
//   h (bf16) rewritten into LDS, c kept in VGPRs, output rows 2w..2w+1 written.
// ---------------------------------------------------------------------------
__global__ __launch_bounds__(256) void lstm_scan(
    const unsigned short* __restrict__ pre, const unsigned short* __restrict__ wh,
    float* __restrict__ G, unsigned int* __restrict__ counter,
    float* __restrict__ out)
{
  __shared__ alignas(16) unsigned short Whl[32][264];    // Wh^T slice (col-major)
  __shared__ alignas(16) unsigned short Hl[64][264];     // h (bf16), full batch
  const int wgid = blockIdx.x;                           // 0..31
  const int g = wgid >> 3, oct = wgid & 7;
  const int tid = threadIdx.x;

  // Stage Wh^T slice: Wh[g][k][oct*32 + c] -> Whl[c][k]
  {
    const unsigned short* s = wh + (size_t)g * 65536 + oct * 32;
    for (int i = 0; i < 4; ++i) {
      const int unit = tid + 256 * i;                    // 1024 units of 8
      const int cb = (unit & 3) * 8, k = unit >> 2;
      us8 v = *(const us8*)&s[(size_t)k * 256 + cb];
#pragma unroll
      for (int r = 0; r < 8; ++r) Whl[cb + r][k] = v[r];
    }
  }
  {
    unsigned short* hp = &Hl[0][0];
    for (int u = tid; u < 64 * 264; u += 256) hp[u] = 0; // h0 = 0
  }
  float c_state[64];
#pragma unroll
  for (int j = 0; j < 64; ++j) c_state[j] = 0.0f;        // c0 = 0
  __syncthreads();

  const int w = tid >> 5, lane = tid & 31;
  const int mt = w >> 1, nt = w & 1;                     // 4 M-tiles x 2 N-tiles
  const int l15 = lane & 15, k8 = ((lane >> 4) & 1) * 8;
  const unsigned short* ap = &Hl[mt * 16 + l15][0];
  const unsigned short* bp = &Whl[nt * 16 + l15][0];
  const int ctile = g * 16 + oct * 2 + nt;
  const int ncol  = oct * 32 + nt * 16 + l15;            // column within gate
  const int hi8r  = ((lane >> 4) & 1) * 8;

  for (int t = 0; t < 2048; ++t) {
    const int p = t & 1;
    // ---- step GEMM: acc = h @ Wh_slice (tile mt x nt) ----
    v8f acc = {};
#pragma unroll
    for (int kt = 0; kt < 8; ++kt) {
      const int ka = kt * 32 + k8;
      v16bf a = CAT16(*(const v8bf*)&ap[ka], *(const v8bf*)&ap[ka + 16]);
      v16bf b = CAT16(*(const v8bf*)&bp[ka], *(const v8bf*)&bp[ka + 16]);
      acc = __builtin_amdgcn_wmma_f32_16x16x32_bf16(false, a, false, b, (short)0, acc, false, false);
    }
    // ---- add pre (bf16 fragment) and store G (fp32) ----
    const size_t pidx = ((((size_t)t * 4 + mt) * 64 + ctile) * 32 + lane) * 8;
    const us8 pv = *(const us8*)&pre[pidx];
    if (t + 1 < 2048)                                    // next timestep fragment
      __builtin_prefetch(&pre[pidx + (size_t)4 * 64 * 32 * 8], 0, 0);
    float* Gp = G + (((size_t)p * 4 + g) << 14);         // 64*256 per gate
#pragma unroll
    for (int r = 0; r < 8; ++r) {
      const int row = mt * 16 + r + hi8r;
      Gp[row * 256 + ncol] = acc[r] + bf2f(pv[r]);
    }
    // ---- global barrier across 32 WGs ----
    __threadfence();
    __syncthreads();
    if (tid == 0) {
      __hip_atomic_fetch_add(counter, 1u, __ATOMIC_RELEASE, __HIP_MEMORY_SCOPE_AGENT);
      const unsigned target = 32u * (unsigned)(t + 1);
      while (__hip_atomic_load(counter, __ATOMIC_ACQUIRE, __HIP_MEMORY_SCOPE_AGENT) < target)
        __builtin_amdgcn_s_sleep(1);
    }
    __syncthreads();
    // ---- elementwise gate fusion (redundant per WG; n = tid, b = j) ----
    const float* Gf = G + (((size_t)p * 4 + 0) << 14);
    const float* Gi = G + (((size_t)p * 4 + 1) << 14);
    const float* Gc = G + (((size_t)p * 4 + 2) << 14);
    const float* Go = G + (((size_t)p * 4 + 3) << 14);
#pragma unroll
    for (int j = 0; j < 64; ++j) {
      const int e = j * 256 + tid;
      const float vf = __hip_atomic_load(&Gf[e], __ATOMIC_RELAXED, __HIP_MEMORY_SCOPE_AGENT);
      const float vi = __hip_atomic_load(&Gi[e], __ATOMIC_RELAXED, __HIP_MEMORY_SCOPE_AGENT);
      const float vc = __hip_atomic_load(&Gc[e], __ATOMIC_RELAXED, __HIP_MEMORY_SCOPE_AGENT);
      const float vo = __hip_atomic_load(&Go[e], __ATOMIC_RELAXED, __HIP_MEMORY_SCOPE_AGENT);
      const float cn = sigf(vf) * c_state[j] + sigf(vi) * tanhf_(vc);
      c_state[j] = cn;
      const float hn = sigf(vo) * tanhf_(cn);
      Hl[j][tid] = f2bf(hn);
      if ((j >> 1) == wgid)                              // this WG owns rows 2w,2w+1
        out[((size_t)j * 2048 + t) * 256 + tid] = hn;
    }
    __syncthreads();                                     // h ready for next GEMM
  }
}

// ---------------------------------------------------------------------------
// Host launcher
// ---------------------------------------------------------------------------
extern "C" void kernel_launch(void* const* d_in, const int* in_sizes, int n_in,
                              void* d_out, int out_size, void* d_ws, size_t ws_size,
                              hipStream_t stream) {
  (void)in_sizes; (void)n_in; (void)out_size; (void)ws_size;
  const float* x  = (const float*)d_in[0];
  const float* Wf = (const float*)d_in[1]; const float* bf_ = (const float*)d_in[2];
  const float* Wi = (const float*)d_in[3]; const float* bi_ = (const float*)d_in[4];
  const float* Wc = (const float*)d_in[5]; const float* bc_ = (const float*)d_in[6];
  const float* Wo = (const float*)d_in[7]; const float* bo_ = (const float*)d_in[8];

  char* ws = (char*)d_ws;
  unsigned int*   counter = (unsigned int*)ws;                           // 256 B
  unsigned short* wx      = (unsigned short*)(ws + 256);                 // 512 KiB
  unsigned short* wh      = wx + (size_t)4 * 256 * 256;                  // 512 KiB
  unsigned short* pre     = wh + (size_t)4 * 256 * 256;                  // 256 MiB
  float*          G       = (float*)(ws + 256 + 2 * 524288 + 268435456); // 512 KiB

  lstm_convw<<<2048, 256, 0, stream>>>(Wf, Wi, Wc, Wo, wx, wh, counter);
  lstm_pregemm<<<dim3(16, 2, 2048), 128, 0, stream>>>(x, wx, bf_, bi_, bc_, bo_, pre);
  lstm_scan<<<32, 256, 0, stream>>>(pre, wh, G, counter, (float*)d_out);
}
